// RNN_custom_36747740184798
// MI455X (gfx1250) — matrix operand order repaired and checked by reading, not compiled, not run on hardware
//
#include <hip/hip_runtime.h>
#include <math.h>
#include <stdint.h>

typedef __attribute__((ext_vector_type(16))) _Float16 v16h;
typedef __attribute__((ext_vector_type(8)))  _Float16 v8h;
typedef __attribute__((ext_vector_type(8)))  float    v8f;

// global-address-space pointer types (keep weight streams on the GLOBAL path:
// flat_load increments both LOADcnt and DScnt and pollutes the ds waits)
typedef const __attribute__((address_space(1))) _Float16* gcptr;
typedef const __attribute__((address_space(1))) v16h*     gv16p;

// ---- problem constants ----
#define B_SZ    256
#define T_STEPS 250
#define DIN     700
#define DIN_P   704           // pad to multiple of 32 (K of wmma f16)
#define KT_X    (DIN_P / 32)  // 22 k-tiles for input GEMM
#define H       256
#define KT_H    (H / 32)      // 8 k-tiles for 256-wide GEMMs
#define NT_H    (H / 16)      // 16 n-tiles (one per wave)
#define DOUT    20
#define DOUT_P  32
#define NT_O    2
#define XSTR    712           // LDS stride (halfs) for x tile: bank-conflict-free
#define XSZ     (16 * XSTR)
#define SSTR    264           // LDS stride (halfs) for spike tiles
#define B_J0    0.01f
#define BETA_A  1.8f

// ---- workspace layout (element counts in _Float16 for weights) ----
constexpr size_t SZ_WI  = (size_t)KT_X * NT_H * 512;   // 180224 halfs
constexpr size_t SZ_WH  = (size_t)KT_H * NT_H * 512;   //  65536 halfs
constexpr size_t SZ_WO  = (size_t)KT_H * NT_O * 512;   //   8192 halfs
constexpr size_t OFF_WI  = 0;
constexpr size_t OFF_WR1 = OFF_WI  + SZ_WI;
constexpr size_t OFF_W12 = OFF_WR1 + SZ_WH;
constexpr size_t OFF_WR2 = OFF_W12 + SZ_WH;
constexpr size_t OFF_WO  = OFF_WR2 + SZ_WH;
constexpr size_t HALF_TOTAL = OFF_WO + SZ_WO;          // 385024 halfs
constexpr size_t PARAM_BYTE_OFF = HALF_TOTAL * 2;      // 770048 B (256-aligned)
// float param layout: [0,256) alpha1 | [256,512) ro1 | [512,768) alpha2 |
// [768,1024) ro2 | [1024,1056) alpha_o(pad) | [1056,1312) b_h1 |
// [1312,1568) b_h2 | [1568,1600) b_o(pad)

// ------------------------------------------------------------------
// Prep: transpose + f16-convert weights into WMMA B-fragment blocks.
// Block (kt, nt) holds a 32(K)x16(N) tile; per lane 16 contiguous halfs:
//   lanes 0-15  -> K = kt*32 + {0..7, 16..23},  N = nt*16 + lane
//   lanes 16-31 -> K = kt*32 + {8..15, 24..31}, N = nt*16 + (lane-16)
// ------------------------------------------------------------------
__global__ void prep_weights(const float* __restrict__ w, _Float16* __restrict__ dst,
                             int KTt, int NTt, int IN, int OUTd) {
  int total = KTt * NTt * 512;
  for (int i = blockIdx.x * blockDim.x + threadIdx.x; i < total;
       i += gridDim.x * blockDim.x) {
    int e    = i & 511;
    int blk  = i >> 9;
    int nt   = blk % NTt;
    int kt   = blk / NTt;
    int lane = e >> 4;
    int j    = e & 15;
    int hf   = lane >> 4;
    int ln   = lane & 15;
    int k    = kt * 32 + hf * 8 + (j & 7) + (j >> 3) * 16;
    int n    = nt * 16 + ln;
    float v  = (k < IN && n < OUTd) ? w[(size_t)n * IN + k] : 0.f;
    dst[i] = (_Float16)v;
  }
}

__global__ void prep_params(const float* __restrict__ tm1, const float* __restrict__ ta1,
                            const float* __restrict__ tm2, const float* __restrict__ ta2,
                            const float* __restrict__ tmo, const float* __restrict__ bh1,
                            const float* __restrict__ bh2, const float* __restrict__ bo,
                            float* __restrict__ p) {
  int i = threadIdx.x + blockIdx.x * blockDim.x;
  if (i < 256) {
    p[i]        = __expf(-1.f / tm1[i]);
    p[256 + i]  = __expf(-1.f / ta1[i]);
    p[512 + i]  = __expf(-1.f / tm2[i]);
    p[768 + i]  = __expf(-1.f / ta2[i]);
    p[1056 + i] = bh1[i];
    p[1312 + i] = bh2[i];
  }
  if (i < 32) {
    p[1024 + i] = (i < DOUT) ? __expf(-1.f / tmo[i]) : 0.f;  // pad alpha=0 -> mem stays 0
    p[1568 + i] = (i < DOUT) ? bo[i] : 0.f;
  }
}

// Compiler barrier on a pointer value: stops LICM from hoisting the weight
// loads out of the 250-step loop (which blew the VGPR file and spilled),
// while keeping the pointer in the GLOBAL address space (global_load_b128).
__device__ __forceinline__ gcptr launder_g(const _Float16* ptr) {
  uintptr_t v = (uintptr_t)ptr;
  asm volatile("" : "+s"(v));
  return (gcptr)v;
}

// A fragment (16x32 f16, ISA layout) from an LDS row-major [16][stride] tile.
__device__ __forceinline__ v16h loadA(const _Float16* base, int stride, int kt,
                                      int hf, int ln) {
  const _Float16* pA = base + ln * stride + kt * 32 + hf * 8;
  v8h lo = *(const v8h*)pA;          // K +0..7  (or +8..15 in hi half-wave)
  v8h hi = *(const v8h*)(pA + 16);   // K +16..23 (or +24..31)
  v16h a;
#pragma unroll
  for (int i = 0; i < 8; ++i) { a[i] = lo[i]; a[i + 8] = hi[i]; }
  return a;
}

// B fragment: 32 bytes per lane, fully coalesced (1 KB/block from L2).
__device__ __forceinline__ v16h loadB(const _Float16* __restrict__ blockBase, int lane) {
  return *(const v16h*)(blockBase + lane * 16);
}
__device__ __forceinline__ v16h loadB_g(gcptr blockBase, int lane) {
  return *(gv16p)(blockBase + lane * 16);
}

// Software-pipelined GEMM: LDS A-tile x streamed global B-fragments,
// double-buffered so fetch of kt+1 overlaps the WMMA for kt.
template <int KT, int NT>
__device__ __forceinline__ void gemm_db(v8f& acc, const _Float16* aLds, int astr,
                                        gcptr bGlob, int nt,
                                        int lane, int hf, int ln) {
  v16h a0 = loadA(aLds, astr, 0, hf, ln);
  v16h b0 = loadB_g(bGlob + (size_t)nt * 512, lane);
#pragma unroll
  for (int kt = 0; kt < KT - 1; ++kt) {
    v16h a1 = loadA(aLds, astr, kt + 1, hf, ln);
    v16h b1 = loadB_g(bGlob + (size_t)((kt + 1) * NT + nt) * 512, lane);
    acc = __builtin_amdgcn_wmma_f32_16x16x32_f16(false, a0, false, b0,
                                                 (short)0, acc, false, false);
    a0 = a1; b0 = b1;
  }
  acc = __builtin_amdgcn_wmma_f32_16x16x32_f16(false, a0, false, b0,
                                               (short)0, acc, false, false);
}

// GEMM against register-resident B fragments (Wr1), A-side double-buffered.
__device__ __forceinline__ void gemm_regB(v8f& acc, const _Float16* aLds, int astr,
                                          const v16h* wreg, int hf, int ln) {
  v16h a0 = loadA(aLds, astr, 0, hf, ln);
#pragma unroll
  for (int kt = 0; kt < KT_H - 1; ++kt) {
    v16h a1 = loadA(aLds, astr, kt + 1, hf, ln);
    acc = __builtin_amdgcn_wmma_f32_16x16x32_f16(false, a0, false, wreg[kt],
                                                 (short)0, acc, false, false);
    a0 = a1;
  }
  acc = __builtin_amdgcn_wmma_f32_16x16x32_f16(false, a0, false, wreg[KT_H - 1],
                                               (short)0, acc, false, false);
}

// Stage one timestep's 16x700 fp32 x-rows into an LDS f16 A-tile (padded).
__device__ __forceinline__ void stage_x(_Float16* dst, const float* __restrict__ x,
                                        int rowbase, int t, int tid) {
  const float* xt = x + (size_t)rowbase * T_STEPS * DIN + (size_t)t * DIN;
#pragma unroll 2
  for (int i = tid; i < 16 * DIN_P; i += 512) {
    int r = i / DIN_P;
    int c = i - r * DIN_P;
    float v = (c < DIN) ? xt[(size_t)r * T_STEPS * DIN + c] : 0.f;
    dst[r * XSTR + c] = (_Float16)v;
  }
}

// Adaptive-LIF update on one 16x16 f32 accumulator (D layout).
// Reads spk(t-1) from LDS at this wave's own columns, writes spk(t).
__device__ __forceinline__ void lif_update(v8f& acc, v8f& mem, v8f& bb,
                                           float alpha, float ro,
                                           _Float16* spkLds, int hf, int ln, int nbase) {
  const float bfac = BETA_A * (1.f - ro);
  const float oma  = 1.f - alpha;
#pragma unroll
  for (int r = 0; r < 8; ++r) {
    int m   = r + hf * 8;
    int idx = m * SSTR + nbase + ln;
    float so = (float)spkLds[idx];                       // spike(t-1)
    bb[r]  = ro * bb[r] + bfac * so;
    mem[r] = mem[r] * alpha - bb[r] * so + oma * acc[r];
    float sn = (mem[r] - bb[r] - B_J0) > 0.f ? 1.f : 0.f;
    spkLds[idx] = (_Float16)sn;                          // spike(t)
  }
}

__launch_bounds__(512, 1)
__global__ void snn_persistent(const float* __restrict__ x,
                               const _Float16* __restrict__ wbuf,
                               const float* __restrict__ p,
                               float* __restrict__ out) {
  __shared__ _Float16 xA[2 * XSZ];        // ping-pong x(t) tiles, f16, padded stride
  __shared__ _Float16 spk1[16 * SSTR];    // h1 spikes (A matrix for rec/fwd GEMMs)
  __shared__ _Float16 spk2[16 * SSTR];    // h2 spikes
  __shared__ float    memo[16 * DOUT_P];  // output membrane staging
  __shared__ float    accS[16 * DOUT];    // softmax accumulator

  const int tid  = threadIdx.x;
  const int lane = tid & 31;
  const int wv   = tid >> 5;       // 16 waves; wave w owns n-tile w of h1 and h2
  const int hf   = lane >> 4;
  const int ln   = lane & 15;
  const int rowbase = blockIdx.x * 16;   // 16 batch rows per workgroup

  for (int i = tid; i < 16 * SSTR; i += 512) {
    spk1[i] = (_Float16)0.f; spk2[i] = (_Float16)0.f;
  }
  for (int i = tid; i < 16 * DOUT; i += 512) accS[i] = 0.f;

  const int n1 = wv * 16 + ln;     // this lane's neuron column
  const float alpha1 = p[n1],       ro1 = p[256 + n1];
  const float alpha2 = p[512 + n1], ro2 = p[768 + n1];
  const float bias1  = p[1056 + n1], bias2 = p[1312 + n1];
  float alphao = 0.f, biaso = 0.f;
  if (wv < 2) { int no = wv * 16 + ln; alphao = p[1024 + no]; biaso = p[1568 + no]; }

  v8f mem1, bb1, mem2, bb2, memO;
#pragma unroll
  for (int r = 0; r < 8; ++r) {
    mem1[r] = 0.f; bb1[r] = B_J0; mem2[r] = 0.f; bb2[r] = B_J0; memO[r] = 0.f;
  }

  // Pin Wr1 (the recurrent matrix on the critical h1 chain) in registers:
  // 8 fragments x 16 halfs = 64 VGPRs, loaded once for all 250 steps.
  v16h wr1[KT_H];
#pragma unroll
  for (int kt = 0; kt < KT_H; ++kt)
    wr1[kt] = loadB(wbuf + OFF_WR1 + (size_t)(kt * NT_H + wv) * 512, lane);

  // prologue: stage x(0) into buffer 0
  stage_x(xA, x, rowbase, 0, tid);
  __syncthreads();

  for (int t = 0; t < T_STEPS; ++t) {
    const _Float16* xCur = xA + (t & 1) * XSZ;
    _Float16*       xNxt = xA + ((t + 1) & 1) * XSZ;

    // Laundered (non-hoistable) weight bases: streamed from L2 every step.
    gcptr WiFt  = launder_g(wbuf + OFF_WI);
    gcptr W12Ft = launder_g(wbuf + OFF_W12);
    gcptr Wr2Ft = launder_g(wbuf + OFF_WR2);
    gcptr WoFt  = launder_g(wbuf + OFF_WO);

    // ---- Phase B: h1_in = x @ Wi^T + spk1(t-1) @ Wr1^T + b1 ----
    v8f acc;
#pragma unroll
    for (int r = 0; r < 8; ++r) acc[r] = bias1;
    gemm_db<KT_X, NT_H>(acc, xCur, XSTR, WiFt, wv, lane, hf, ln);

    // stage x(t+1) into the other buffer: global loads fly under this step's
    // compute; end-of-step barriers order the LDS writes vs. next step's reads.
    if (t + 1 < T_STEPS) stage_x(xNxt, x, rowbase, t + 1, tid);
    // and pull x(t+2) toward L2 (global_prefetch_b8)
    if (t + 2 < T_STEPS && tid < 176) {
      const float* xn = x + (size_t)rowbase * T_STEPS * DIN + (size_t)(t + 2) * DIN;
      int r = tid / 11, cl = tid - r * 11;
      __builtin_prefetch(xn + (size_t)r * T_STEPS * DIN + cl * 64, 0, 1);
    }

    gemm_regB(acc, spk1, SSTR, wr1, hf, ln);
    __syncthreads();   // everyone done reading spk1(t-1)
    lif_update(acc, mem1, bb1, alpha1, ro1, spk1, hf, ln, wv * 16);
    __syncthreads();   // spk1(t) visible

    // ---- Phase C: h2_in = spk1(t) @ W12^T + spk2(t-1) @ Wr2^T + b2 ----
#pragma unroll
    for (int r = 0; r < 8; ++r) acc[r] = bias2;
    gemm_db<KT_H, NT_H>(acc, spk1, SSTR, W12Ft, wv, lane, hf, ln);
    gemm_db<KT_H, NT_H>(acc, spk2, SSTR, Wr2Ft, wv, lane, hf, ln);
    __syncthreads();   // everyone done reading spk2(t-1)
    lif_update(acc, mem2, bb2, alpha2, ro2, spk2, hf, ln, wv * 16);
    __syncthreads();   // spk2(t) visible

    // ---- Phase D: readout (waves 0-1; wave-uniform branch, EXEC all-ones) ----
    if (wv < 2) {
#pragma unroll
      for (int r = 0; r < 8; ++r) acc[r] = biaso;
      gemm_db<KT_H, NT_O>(acc, spk2, SSTR, WoFt, wv, lane, hf, ln);
#pragma unroll
      for (int r = 0; r < 8; ++r) {
        memO[r] = memO[r] * alphao + (1.f - alphao) * acc[r];
        memo[(r + hf * 8) * DOUT_P + wv * 16 + ln] = memO[r];
      }
    }
    __syncthreads();

    // ---- Phase E: softmax over 20 classes, accumulate (16 rows) ----
    if (tid < 16) {
      float v[DOUT];
      float mx = -1e30f;
#pragma unroll
      for (int j = 0; j < DOUT; ++j) { v[j] = memo[tid * DOUT_P + j]; mx = fmaxf(mx, v[j]); }
      float s = 0.f;
#pragma unroll
      for (int j = 0; j < DOUT; ++j) { float e = __expf(v[j] - mx); v[j] = e; s += e; }
      float inv = 1.f / s;
#pragma unroll
      for (int j = 0; j < DOUT; ++j) accS[tid * DOUT + j] += v[j] * inv;
    }
    __syncthreads();
  }

  // ---- write accumulated softmax: rows are contiguous in out ----
  for (int i = tid; i < 16 * DOUT; i += 512)
    out[(size_t)rowbase * DOUT + i] = accS[i];
}

extern "C" void kernel_launch(void* const* d_in, const int* in_sizes, int n_in,
                              void* d_out, int out_size, void* d_ws, size_t ws_size,
                              hipStream_t stream) {
  const float* x        = (const float*)d_in[0];
  const float* w_i2h1   = (const float*)d_in[1];
  const float* w_h12h1  = (const float*)d_in[2];
  const float* w_h12h2  = (const float*)d_in[3];
  const float* w_h22h2  = (const float*)d_in[4];
  const float* w_h2o    = (const float*)d_in[5];
  const float* b_h1     = (const float*)d_in[6];
  const float* b_h2     = (const float*)d_in[7];
  const float* b_o      = (const float*)d_in[8];
  const float* tau_adp1 = (const float*)d_in[9];
  const float* tau_adp2 = (const float*)d_in[10];
  const float* tau_m1   = (const float*)d_in[11];
  const float* tau_m2   = (const float*)d_in[12];
  const float* tau_mo   = (const float*)d_in[13];

  _Float16* wbuf = (_Float16*)d_ws;
  float*    prm  = (float*)((char*)d_ws + PARAM_BYTE_OFF);

  prep_weights<<<256, 256, 0, stream>>>(w_i2h1,  wbuf + OFF_WI,  KT_X, NT_H, DIN, H);
  prep_weights<<<64,  256, 0, stream>>>(w_h12h1, wbuf + OFF_WR1, KT_H, NT_H, H,   H);
  prep_weights<<<64,  256, 0, stream>>>(w_h12h2, wbuf + OFF_W12, KT_H, NT_H, H,   H);
  prep_weights<<<64,  256, 0, stream>>>(w_h22h2, wbuf + OFF_WR2, KT_H, NT_H, H,   H);
  prep_weights<<<32,  256, 0, stream>>>(w_h2o,   wbuf + OFF_WO,  KT_H, NT_O, H,   DOUT);
  prep_params<<<1, 256, 0, stream>>>(tau_m1, tau_adp1, tau_m2, tau_adp2, tau_mo,
                                     b_h1, b_h2, b_o, prm);

  snn_persistent<<<B_SZ / 16, 512, 0, stream>>>(x, wbuf, prm, (float*)d_out);
}
